// RNN_15281493639559
// MI455X (gfx1250) — matrix operand order repaired and checked by reading, not compiled, not run on hardware
//
#include <hip/hip_runtime.h>

// Persistent-kernel RNN for MI455X (gfx1250, wave32, WMMA).
// 64 workgroups x 256 threads (8 waves). Workgroup wg owns output columns
// [16*wg, 16*wg+16). Wave w owns rows [16*w, 16*w+16): one 16x16 WMMA C tile.
// x1 state lives in 8 VGPRs/lane (exactly the WMMA f32 C layout).
// Masked W_hh slice (16 x 1024, bf16) preloaded to LDS once, reused 513 steps.

#define TT    512
#define BB    128
#define NN    1024
#define XDIM  7
#define NIN_  3
#define NOUT_ 2
#define ALPHA 0.1f
#define DT_   0.01f
#define NWG   64      // NN / 16
#define NTHR  256

typedef __attribute__((ext_vector_type(16))) __bf16    v16bf;
typedef __attribute__((ext_vector_type(8)))  float     v8f;
typedef __attribute__((ext_vector_type(4)))  unsigned  u32x4;

union Frag { u32x4 q[2]; v16bf v; };

__device__ __forceinline__ unsigned short f2bf(float f) {
  unsigned u = __float_as_uint(f);
  return (unsigned short)((u + 0x7FFFu + ((u >> 16) & 1u)) >> 16);
}

__device__ __forceinline__ void grid_barrier(unsigned* cnt, unsigned target) {
  __threadfence();
  __syncthreads();
  if (threadIdx.x == 0) {
    __hip_atomic_fetch_add(cnt, 1u, __ATOMIC_ACQ_REL, __HIP_MEMORY_SCOPE_AGENT);
    while (__hip_atomic_load(cnt, __ATOMIC_ACQUIRE, __HIP_MEMORY_SCOPE_AGENT) < target)
      __builtin_amdgcn_s_sleep(2);
  }
  __syncthreads();
  __threadfence();   // acquire: make other blocks' stores visible to all lanes
}

__global__ __launch_bounds__(NTHR, 1)
void rnn_persistent(const float* __restrict__ X,        // [T,B,7]
                    const float* __restrict__ Xpert,    // [T,B,2]
                    const float* __restrict__ popto,    // [T,B,N]
                    const float* __restrict__ hidden0,  // [B,N]
                    const float* __restrict__ W_hh,     // [N,N]
                    const float* __restrict__ W_ih,     // [N,3]
                    const float* __restrict__ W_fb,     // [N,2]
                    const float* __restrict__ b_fb,     // [N]
                    const float* __restrict__ W_out,    // [2,N]
                    const float* __restrict__ b_out,    // [2]
                    const float* __restrict__ mask_fb,  // [N,2]
                    const float* __restrict__ mask_rec, // [N,N]
                    float* __restrict__ out_poserr,     // [T,B,2]
                    float* __restrict__ out_hidden,     // [T,B,N]
                    unsigned short* __restrict__ rbuf0, // [B,N] bf16 ping
                    unsigned short* __restrict__ rbuf1, // [B,N] bf16 pong
                    float* __restrict__ vtp,            // 2 x [NWG,B,2]
                    unsigned* __restrict__ barrier_cnt)
{
  __shared__ unsigned short Wsl[16 * NN];       // masked W_hh rows n0..n0+15, bf16
  __shared__ float Wout_sl[NOUT_][16];
  __shared__ float Wih_sl[NIN_][16];
  __shared__ float Wfb_sl[NOUT_][16];
  __shared__ float bfb_sl[16];
  __shared__ float v_sh[BB * NOUT_];            // current velocity (replicated)

  const int wg   = blockIdx.x;
  const int n0   = wg * 16;
  const int tid  = threadIdx.x;
  const int wv   = tid >> 5;
  const int lane = tid & 31;
  const int colL = lane & 15;     // C-tile column within slice, also A-row low bits
  const int hh   = lane >> 4;     // lane half (affects K offsets & C row +8)

  // ---- one-time preload: masked recurrent weights -> LDS bf16 -------------
  for (int idx = tid; idx < 16 * NN; idx += NTHR) {
    int n = idx >> 10, k = idx & (NN - 1);
    float w = W_hh[(size_t)(n0 + n) * NN + k] * mask_rec[(size_t)(n0 + n) * NN + k];
    Wsl[idx] = f2bf(w);
  }
  if (tid < 16) {
    bfb_sl[tid] = b_fb[n0 + tid];
    for (int o = 0; o < NOUT_; ++o) {
      Wfb_sl[o][tid]  = W_fb[(n0 + tid) * NOUT_ + o] * mask_fb[(n0 + tid) * NOUT_ + o];
      Wout_sl[o][tid] = W_out[o * NN + n0 + tid];
    }
    for (int i = 0; i < NIN_; ++i) Wih_sl[i][tid] = W_ih[(n0 + tid) * NIN_ + i];
  }
  __syncthreads();

  // vt-partial helper: reduce r[g]*Wout over this wave's 16 columns per row.
  auto vt_partial = [&](const float* rr, float* vb) {
#pragma unroll
    for (int g = 0; g < 8; ++g) {
      float s0 = rr[g] * Wout_sl[0][colL];
      float s1 = rr[g] * Wout_sl[1][colL];
#pragma unroll
      for (int m = 1; m < 16; m <<= 1) {
        s0 += __shfl_xor(s0, m, 32);
        s1 += __shfl_xor(s1, m, 32);
      }
      if (colL == 0) {
        int b = wv * 16 + g + 8 * hh;
        vb[((size_t)wg * BB + b) * NOUT_ + 0] = s0;
        vb[((size_t)wg * BB + b) * NOUT_ + 1] = s1;
      }
    }
  };

  // ---- init: x = hidden0, r = relu(x), v = r @ Wout^T + b_out -------------
  float x[8], rn[8];
#pragma unroll
  for (int g = 0; g < 8; ++g) {
    int b = wv * 16 + g + 8 * hh;
    x[g]  = hidden0[(size_t)b * NN + n0 + colL];
    rn[g] = fmaxf(x[g], 0.f);
    rbuf0[(size_t)b * NN + n0 + colL] = f2bf(rn[g]);
  }
  vt_partial(rn, vtp);            // phase 0 -> vtp buffer 0
  unsigned target = NWG;
  grid_barrier(barrier_cnt, target);
  {
    int b = tid >> 1, o = tid & 1;
    float s = 0.f;
#pragma unroll 8
    for (int w2 = 0; w2 < NWG; ++w2) s += vtp[((size_t)w2 * BB + b) * NOUT_ + o];
    v_sh[b * NOUT_ + o] = s + b_out[o];
  }
  __syncthreads();

  // ---- 513 sequential steps: iter 0 = prologue (not stored), iter i>=1 -> t=i-1
  for (int iter = 0; iter <= TT; ++iter) {
    const unsigned short* rprev = (iter & 1) ? rbuf1 : rbuf0;
    unsigned short*       rcur  = (iter & 1) ? rbuf0 : rbuf1;
    float* vtpb = vtp + (((iter + 1) & 1) ? (size_t)NWG * BB * NOUT_ : 0);
    const int  t     = (iter == 0) ? 0 : iter - 1;
    const bool store = iter > 0;
    const bool usefb = iter >= 2;   // prologue and scan t=0 use zero feedback

    // -------- pre = r_prev @ Whh_m^T : 16x16 tile, K=1024, bf16 WMMA ------
    v8f c0 = {}, c1 = {};
    const unsigned short* abase = rprev + (size_t)(wv * 16 + colL) * NN + 8 * hh;
    const unsigned short* bbase = Wsl + colL * NN + 8 * hh;
#pragma unroll 4
    for (int kc = 0; kc < 32; kc += 2) {
      {
        Frag a, bmat;
        const u32x4* pa = (const u32x4*)(abase + kc * 32);
        a.q[0] = pa[0]; a.q[1] = pa[2];                 // K {0..7,16..23} of chunk
        const u32x4* pb = (const u32x4*)(bbase + kc * 32);
        bmat.q[0] = pb[0]; bmat.q[1] = pb[2];           // K {0..7,16..23} of chunk, ds_load_b128
        c0 = __builtin_amdgcn_wmma_f32_16x16x32_bf16(false, a.v, false, bmat.v,
                                                     (short)0, c0, false, false);
      }
      {
        Frag a, bmat;
        const u32x4* pa = (const u32x4*)(abase + (kc + 1) * 32);
        a.q[0] = pa[0]; a.q[1] = pa[2];
        const u32x4* pb = (const u32x4*)(bbase + (kc + 1) * 32);
        bmat.q[0] = pb[0]; bmat.q[1] = pb[2];
        c1 = __builtin_amdgcn_wmma_f32_16x16x32_bf16(false, a.v, false, bmat.v,
                                                     (short)0, c1, false, false);
      }
    }

    // -------- elementwise: small terms + leaky update + relu --------------
#pragma unroll
    for (int g = 0; g < 8; ++g) {
      int b = wv * 16 + g + 8 * hh;
      int n = n0 + colL;
      const float* xrow = X + ((size_t)t * BB + b) * XDIM;
      float pre = c0[g] + c1[g] + bfb_sl[colL]
                + __builtin_nontemporal_load(popto + ((size_t)t * BB + b) * NN + n)
                + xrow[0] * Wih_sl[0][colL]
                + xrow[1] * Wih_sl[1][colL]
                + xrow[2] * Wih_sl[2][colL];
      if (usefb)
        pre += v_sh[b * NOUT_ + 0] * Wfb_sl[0][colL]
             + v_sh[b * NOUT_ + 1] * Wfb_sl[1][colL];
      x[g]  = (1.f - ALPHA) * x[g] + ALPHA * pre;
      rn[g] = fmaxf(x[g], 0.f);
      rcur[(size_t)b * NN + n] = f2bf(rn[g]);
      if (store)
        __builtin_nontemporal_store(rn[g], out_hidden + ((size_t)t * BB + b) * NN + n);
    }

    vt_partial(rn, vtpb);

    target += NWG;
    grid_barrier(barrier_cnt, target);

    // -------- velocity update (replicated, deterministic) -----------------
    {
      int b = tid >> 1, o = tid & 1;
      float s = 0.f;
#pragma unroll 8
      for (int w2 = 0; w2 < NWG; ++w2) s += vtpb[((size_t)w2 * BB + b) * NOUT_ + o];
      float vt   = s + b_out[o] + Xpert[((size_t)t * BB + b) * NOUT_ + o];
      float xin  = X[((size_t)t * BB + b) * XDIM + 3 + o];
      float vnew = v_sh[b * NOUT_ + o] + DT_ * (xin - vt);
      if (store && wg == 0)
        out_poserr[((size_t)t * BB + b) * NOUT_ + o] = vnew;
      v_sh[b * NOUT_ + o] = vnew;
    }
    __syncthreads();
  }
}

extern "C" void kernel_launch(void* const* d_in, const int* in_sizes, int n_in,
                              void* d_out, int out_size, void* d_ws, size_t ws_size,
                              hipStream_t stream) {
  (void)in_sizes; (void)n_in; (void)out_size; (void)ws_size;
  const float* X        = (const float*)d_in[0];
  const float* Xpert    = (const float*)d_in[1];
  const float* popto    = (const float*)d_in[2];
  const float* hidden0  = (const float*)d_in[3];
  const float* W_hh     = (const float*)d_in[4];
  const float* W_ih     = (const float*)d_in[5];
  const float* W_fb     = (const float*)d_in[6];
  const float* b_fb     = (const float*)d_in[7];
  const float* W_out    = (const float*)d_in[8];
  const float* b_out    = (const float*)d_in[9];
  const float* mask_fb  = (const float*)d_in[10];
  const float* mask_rec = (const float*)d_in[11];

  float* out_poserr = (float*)d_out;                                  // [T,B,2]
  float* out_hidden = out_poserr + (size_t)TT * BB * NOUT_;           // [T,B,N]

  char* ws = (char*)d_ws;
  const size_t rbytes = (size_t)BB * NN * sizeof(unsigned short);     // 256 KB
  unsigned short* rbuf0 = (unsigned short*)ws;
  unsigned short* rbuf1 = (unsigned short*)(ws + rbytes);
  float* vtp = (float*)(ws + 2 * rbytes);                             // 2 x 64 KB
  unsigned* cnt = (unsigned*)(ws + 2 * rbytes +
                              2 * (size_t)NWG * BB * NOUT_ * sizeof(float));

  hipMemsetAsync(cnt, 0, 64, stream);   // graph-capture-legal counter reset
  rnn_persistent<<<NWG, NTHR, 0, stream>>>(X, Xpert, popto, hidden0, W_hh, W_ih,
                                           W_fb, b_fb, W_out, b_out, mask_fb,
                                           mask_rec, out_poserr, out_hidden,
                                           rbuf0, rbuf1, vtp, cnt);
}
